// FusionAfterBEVSEDirect_23072564314115
// MI455X (gfx1250) — compile-verified, matrix operand order checked by reading, not run on hardware
//
#include <hip/hip_runtime.h>
#include <hip/hip_bf16.h>
#include <stdint.h>

// ---------------------------------------------------------------------------
// MI455X (gfx1250) implementation. wave32; all matrix math via
// v_wmma_f32_16x16x32_bf16 (bf16 A/B fragments assembled with b128 loads per
// the CDNA5 ISA 16-bit operand layout, f32 accumulation).
// ---------------------------------------------------------------------------

typedef unsigned short u16;
typedef __attribute__((ext_vector_type(8))) float v8f;

#if defined(__HIP_DEVICE_COMPILE__)
#if __has_builtin(__builtin_amdgcn_wmma_f32_16x16x32_bf16)
#define WMMA_BF16 1
typedef __attribute__((ext_vector_type(16))) __bf16 v16bf;
#endif
#endif

struct Frag { uint4 lo, hi; };

__device__ __forceinline__ uint4 u4zero() { uint4 z; z.x = 0u; z.y = 0u; z.z = 0u; z.w = 0u; return z; }
__device__ __forceinline__ v8f v8zero() { v8f z = {0.f,0.f,0.f,0.f,0.f,0.f,0.f,0.f}; return z; }
__device__ __forceinline__ Frag fragzero() { Frag f; f.lo = u4zero(); f.hi = u4zero(); return f; }

__device__ __forceinline__ v8f wmma16(Frag a, Frag b, v8f c) {
#if defined(WMMA_BF16)
  union U { Frag f; v16bf v; };
  U ua; ua.f = a;
  U ub; ub.f = b;
  return __builtin_amdgcn_wmma_f32_16x16x32_bf16(false, ua.v, false, ub.v, (short)0, c, false, false);
#else
  (void)a; (void)b; return c; // host-pass stub
#endif
}

// bf16 storage conversion (round-to-nearest-even)
__device__ __forceinline__ u16 f2b(float x) {
  union { float f; uint32_t u; } v; v.f = x;
  uint32_t r = v.u + 0x7fffu + ((v.u >> 16) & 1u);
  return (u16)(r >> 16);
}
__device__ __forceinline__ float b2f(u16 h) {
  union { uint32_t u; float f; } v; v.u = ((uint32_t)h) << 16;
  return v.f;
}

// Load a 16x32 bf16 tile into WMMA A/B register layout (ISA 7.12.2):
// lane r=l&15 is the row; lanes 0-15 take K 0-7 / 16-23, lanes 16-31 take
// K 8-15 / 24-31 -> exactly two 16-byte loads per lane.
__device__ __forceinline__ Frag ldtile(const u16* base, int stride, int lane) {
  int r = lane & 15, hb = (lane >> 4) * 8;
  const u16* p = base + r * stride + hb;
  Frag f;
  f.lo = *(const uint4*)(p);
  f.hi = *(const uint4*)(p + 16);
  return f;
}

// ------------------------------- constants ---------------------------------
#define BB   2
#define CI   256
#define CR   128
#define HB   200
#define WBD  200
#define CA   128
#define NHD  8
#define DH   16
#define KTOP 100
#define PIX  80000      // BB*HB*WBD
#define ROWS 400        // BB*HB
#define HW   40000
#define CF   384        // CR+CI
#define KCNV 3456       // CF*9

// ------------------------- generic tiled GEMM ------------------------------
// out[m][n] = A[m][:] . W[n][:] (+bias) (relu?) (+resid), M%16==0,
// one wave per 16xNT*16 strip.
template<int NT, bool RELU>
__global__ __launch_bounds__(32)
void gemm16(const u16* __restrict__ A, int lda,
            const u16* __restrict__ W, int K,
            const float* __restrict__ bias,
            const float* __restrict__ resid, int ldr,
            float* __restrict__ outF, int ldf,
            u16* __restrict__ outB, int ldbo)
{
  int lane = threadIdx.x;
  int m0 = blockIdx.x * 16;
  int n0 = blockIdx.y * (NT * 16);
  v8f acc[NT];
#pragma unroll
  for (int t = 0; t < NT; ++t) acc[t] = v8zero();
  const u16* Arow = A + m0 * lda;
  for (int k = 0; k < K; k += 32) {
    Frag a = ldtile(Arow + k, lda, lane);
#pragma unroll
    for (int t = 0; t < NT; ++t) {
      Frag b = ldtile(W + (n0 + t * 16) * K + k, K, lane);
      acc[t] = wmma16(a, b, acc[t]);
    }
  }
  int nl = lane & 15, hi = lane >> 4;
#pragma unroll
  for (int t = 0; t < NT; ++t) {
    int n = n0 + t * 16 + nl;
    float bv = bias ? bias[n] : 0.f;
#pragma unroll
    for (int v = 0; v < 8; ++v) {
      int m = m0 + v + 8 * hi;
      float val = acc[t][v] + bv;
      if (RELU) val = fmaxf(val, 0.f);
      if (resid) val += resid[m * ldr + n];
      if (outF) outF[m * ldf + n] = val;
      if (outB) outB[m * ldbo + n] = f2b(val);
    }
  }
}

// ------------------------ small elementwise kernels ------------------------
__global__ void cvt_k(const float* __restrict__ in, u16* __restrict__ out, int n) {
  int i = blockIdx.x * blockDim.x + threadIdx.x;
  if (i < n) out[i] = f2b(in[i]);
}

// fuse_w (O,I,3,3) -> [O][tap*CF+ch] bf16 (K = tap-major, channel-contiguous)
__global__ void convw_k(const float* __restrict__ fw, u16* __restrict__ wb) {
  int i = blockIdx.x * blockDim.x + threadIdx.x;
  if (i >= CF * KCNV) return;
  int o = i / KCNV, r = i % KCNV, tap = r / CF, ch = r % CF;
  wb[i] = f2b(fw[(o * CF + ch) * 9 + tap]);
}

// NCHW radar+image -> pixel-major bf16 gin_pm[p][384]
__global__ void pm_k(const float* __restrict__ radar, const float* __restrict__ image,
                     u16* __restrict__ gin) {
  int i = blockIdx.x * blockDim.x + threadIdx.x;
  if (i >= BB * CF * HW) return;
  int b = i / (CF * HW);
  int r = i % (CF * HW);
  int c = r / HW;
  int pix = r % HW;
  float v = (c < CR) ? radar[(b * CR + c) * HW + pix]
                     : image[(b * CI + (c - CR)) * HW + pix];
  gin[(b * HW + pix) * CF + c] = f2b(v);
}

__global__ void score2_k(const u16* __restrict__ hs, const float* __restrict__ w2,
                         const float* __restrict__ b2, float* __restrict__ score) {
  int p = blockIdx.x * blockDim.x + threadIdx.x;
  if (p >= PIX) return;
  float s = b2[0];
  for (int c = 0; c < 64; ++c) s += b2f(hs[p * 64 + c]) * w2[c];
  score[p] = s;
}

// rank-count top-k per row (stable tie-break = lower index wins, like lax.top_k)
__global__ void topk_k(const float* __restrict__ score, int* __restrict__ idx) {
  __shared__ float s[WBD];
  int row = blockIdx.x;
  for (int i = threadIdx.x; i < WBD; i += blockDim.x) s[i] = score[row * WBD + i];
  __syncthreads();
  for (int i = threadIdx.x; i < WBD; i += blockDim.x) {
    float v = s[i];
    int r = 0;
    for (int j = 0; j < WBD; ++j) {
      float u = s[j];
      r += (u > v) || (u == v && j < i);
    }
    if (r < KTOP) idx[row * KTOP + r] = i;
  }
}

__global__ void ln_rows_k(const float* __restrict__ in, const float* __restrict__ g,
                          const float* __restrict__ b, u16* __restrict__ out, int M) {
  int p = blockIdx.x * blockDim.x + threadIdx.x;
  if (p >= M) return;
  const float* row = in + p * CA;
  float s = 0.f, s2 = 0.f;
  for (int c = 0; c < CA; ++c) { float v = row[c]; s += v; s2 += v * v; }
  float mean = s * (1.f / CA);
  float var = s2 * (1.f / CA) - mean * mean;
  float inv = rsqrtf(var + 1e-5f);
  for (int c = 0; c < CA; ++c) out[p * CA + c] = f2b((row[c] - mean) * inv * g[c] + b[c]);
}

// gather img_seq at top-k indices + layernorm -> kvln[row*K+kk][CA]
__global__ void lnkv_k(const float* __restrict__ imgseq, const int* __restrict__ idx,
                       const float* __restrict__ g, const float* __restrict__ b,
                       u16* __restrict__ out) {
  int j = blockIdx.x * blockDim.x + threadIdx.x;
  if (j >= ROWS * KTOP) return;
  int row = j / KTOP, kk = j % KTOP;
  int w = idx[row * KTOP + kk];
  const float* src = imgseq + (row * WBD + w) * CA;
  float s = 0.f, s2 = 0.f;
  for (int c = 0; c < CA; ++c) { float v = src[c]; s += v; s2 += v * v; }
  float mean = s * (1.f / CA);
  float var = s2 * (1.f / CA) - mean * mean;
  float inv = rsqrtf(var + 1e-5f);
  for (int c = 0; c < CA; ++c) out[j * CA + c] = f2b((src[c] - mean) * inv * g[c] + b[c]);
}

// ------------------------------- attention ---------------------------------
// one wave per (row, head): QK^T and att*V via WMMA, softmax in LDS.
__global__ __launch_bounds__(32)
void attn_k(const u16* __restrict__ qh, const u16* __restrict__ kh,
            const u16* __restrict__ vh, u16* __restrict__ obuf)
{
  __shared__ __align__(16) u16 khs[112 * 16];   // [k][d], rows>=100 zero
  __shared__ __align__(16) u16 vts[16 * 128];   // [d][k], cols>=100 zero
  __shared__ __align__(16) float sc[16 * 112];  // scores
  __shared__ __align__(16) u16 att[16 * 128];   // softmaxed, cols>=100 zero
  int lane = threadIdx.x;
  int row = blockIdx.x >> 3;
  int head = blockIdx.x & 7;
  int off = head * DH;
  for (int i = lane; i < 112 * 16; i += 32) khs[i] = 0;
  for (int i = lane; i < 16 * 128; i += 32) { vts[i] = 0; att[i] = 0; }
  __syncthreads();
  for (int kk = lane; kk < KTOP; kk += 32) {
    const u16* kr = kh + (row * KTOP + kk) * CA + off;
    const u16* vr = vh + (row * KTOP + kk) * CA + off;
    for (int d = 0; d < DH; ++d) {
      khs[kk * 16 + d] = kr[d];
      vts[d * 128 + kk] = vr[d];
    }
  }
  __syncthreads();
  int nl = lane & 15, hi = lane >> 4;
  for (int qt = 0; qt < 13; ++qt) {
    int q0 = qt * 16;
    int qrow = q0 + nl; if (qrow > WBD - 1) qrow = WBD - 1;   // clamp tail
    Frag aq;
    aq.lo = *(const uint4*)(qh + (row * WBD + qrow) * CA + off + hi * 8);
    aq.hi = u4zero();                                        // K 16..31 = 0
    for (int kt = 0; kt < 7; ++kt) {
      Frag bk;
      bk.lo = *(const uint4*)(&khs[(kt * 16 + nl) * 16 + hi * 8]);
      bk.hi = u4zero();
      v8f s = v8zero();
      s = wmma16(aq, bk, s);
#pragma unroll
      for (int v = 0; v < 8; ++v)
        sc[(v + 8 * hi) * 112 + kt * 16 + nl] = s[v] * 0.25f; // 1/sqrt(DH)
    }
    __syncthreads();
    if (lane < 16) {
      float mx = -3.0e38f;
      for (int k2 = 0; k2 < KTOP; ++k2) mx = fmaxf(mx, sc[lane * 112 + k2]);
      float sum = 0.f;
      for (int k2 = 0; k2 < KTOP; ++k2) {
        float e = __expf(sc[lane * 112 + k2] - mx);
        sc[lane * 112 + k2] = e; sum += e;
      }
      float inv = 1.f / sum;
      for (int k2 = 0; k2 < KTOP; ++k2) att[lane * 128 + k2] = f2b(sc[lane * 112 + k2] * inv);
    }
    __syncthreads();
    v8f o = v8zero();
    for (int ks = 0; ks < 4; ++ks) {
      Frag a = ldtile(&att[ks * 32], 128, lane);
      Frag b = ldtile(&vts[ks * 32], 128, lane);
      o = wmma16(a, b, o);
    }
#pragma unroll
    for (int v = 0; v < 8; ++v) {
      int m = v + 8 * hi;
      int q = q0 + m;
      if (q < WBD) obuf[(row * WBD + q) * CA + off + nl] = f2b(o[v]);
    }
    __syncthreads();
  }
}

// ------------------------------ fused FFN ----------------------------------
// per 16-pixel tile: ln2 -> (128->512 relu) -> (512->128) -> += x; emits bf16 x.
__global__ __launch_bounds__(32)
void ffn_k(float* __restrict__ x, u16* __restrict__ xbf,
           const float* __restrict__ g, const float* __restrict__ bb,
           const u16* __restrict__ w1, const float* __restrict__ b1,
           const u16* __restrict__ w2, const float* __restrict__ b2)
{
  __shared__ __align__(16) u16 sxln[16 * 128];
  __shared__ __align__(16) u16 sh[16 * 512];
  int lane = threadIdx.x;
  int m0 = blockIdx.x * 16;
  if (lane < 16) {
    const float* row = x + (m0 + lane) * CA;
    float s = 0.f, s2 = 0.f;
    for (int c = 0; c < CA; ++c) { float v = row[c]; s += v; s2 += v * v; }
    float mean = s * (1.f / CA);
    float var = s2 * (1.f / CA) - mean * mean;
    float inv = rsqrtf(var + 1e-5f);
    for (int c = 0; c < CA; ++c)
      sxln[lane * 128 + c] = f2b((row[c] - mean) * inv * g[c] + bb[c]);
  }
  __syncthreads();
  int nl = lane & 15, hi = lane >> 4;
  for (int chunk = 0; chunk < 4; ++chunk) {
    v8f acc[8];
#pragma unroll
    for (int t = 0; t < 8; ++t) acc[t] = v8zero();
    for (int k = 0; k < 128; k += 32) {
      Frag a = ldtile(&sxln[k], 128, lane);
#pragma unroll
      for (int t = 0; t < 8; ++t) {
        Frag b = ldtile(w1 + (chunk * 128 + t * 16) * 128 + k, 128, lane);
        acc[t] = wmma16(a, b, acc[t]);
      }
    }
#pragma unroll
    for (int t = 0; t < 8; ++t) {
      int n = chunk * 128 + t * 16 + nl;
      float bv = b1[n];
#pragma unroll
      for (int v = 0; v < 8; ++v) {
        int m = v + 8 * hi;
        sh[m * 512 + n] = f2b(fmaxf(acc[t][v] + bv, 0.f));
      }
    }
    __syncthreads();
  }
  v8f acc[8];
#pragma unroll
  for (int t = 0; t < 8; ++t) acc[t] = v8zero();
  for (int k = 0; k < 512; k += 32) {
    Frag a = ldtile(&sh[k], 512, lane);
#pragma unroll
    for (int t = 0; t < 8; ++t) {
      Frag b = ldtile(w2 + (t * 16) * 512 + k, 512, lane);
      acc[t] = wmma16(a, b, acc[t]);
    }
  }
#pragma unroll
  for (int t = 0; t < 8; ++t) {
    int n = t * 16 + nl;
    float bv = b2[n];
#pragma unroll
    for (int v = 0; v < 8; ++v) {
      int m = m0 + v + 8 * hi;
      float val = acc[t][v] + bv + x[m * CA + n];
      x[m * CA + n] = val;
      xbf[m * CA + n] = f2b(val);
    }
  }
}

// radar_enhanced -> fi_pm[p][0:128] bf16 (sigmoid gate * row mask * gamma * delta)
__global__ void enhance_k(const float* __restrict__ radar,
                          const float* __restrict__ gatep,
                          const float* __restrict__ delta,
                          const float* __restrict__ gamma,
                          u16* __restrict__ fi) {
  int p = blockIdx.x * blockDim.x + threadIdx.x;
  if (p >= PIX) return;
  int b = p / HW, r = p % HW, hh = r / WBD, w = r % WBD;
  float y = (float)hh / (float)(HB - 1);
  float rmask = fmaxf(1.f - 0.7f * y, 0.3f);
  float gm = gamma[0];
  for (int c = 0; c < CR; ++c) {
    float rv = radar[((b * CR + c) * HB + hh) * WBD + w];
    float gate = rmask / (1.f + __expf(-gatep[p * CR + c]));
    fi[p * CF + c] = f2b(rv + gm * gate * delta[p * CR + c]);
  }
}

__global__ void copyimg_k(const u16* __restrict__ gin, u16* __restrict__ fi) {
  int i = blockIdx.x * blockDim.x + threadIdx.x;
  if (i >= PIX * CI) return;
  int p = i >> 8, c = i & 255;
  fi[p * CF + CR + c] = gin[p * CF + CR + c];
}

// --------------------------- 3x3 conv (implicit GEMM) ----------------------
// one wave: 16 pixels x 128 out-channels, K = 9 taps x 384 ch, bf16 WMMA.
__global__ __launch_bounds__(32)
void conv_k(const u16* __restrict__ fi, const u16* __restrict__ wconv,
            float* __restrict__ out)
{
  __shared__ __align__(16) float sm[128 * 17];
  int lane = threadIdx.x;
  int mt = blockIdx.x;
  int b = mt / (HB * 13);
  int rem = mt % (HB * 13);
  int h = rem / 13;
  int wt = rem % 13;
  int w0 = wt * 16;
  int n0 = blockIdx.y * 128;
  int nl = lane & 15, hi = lane >> 4;
  v8f acc[8];
#pragma unroll
  for (int t = 0; t < 8; ++t) acc[t] = v8zero();
  for (int tap = 0; tap < 9; ++tap) {
    int ky = tap / 3 - 1, kx = tap % 3 - 1;
    int hp = h + ky;
    if (hp < 0 || hp >= HB) continue;           // uniform over wave
    int wp = w0 + kx + nl;
    bool ok = (wp >= 0) && (wp < WBD);          // per-lane SAME padding
    const u16* arow = nullptr;
    if (ok) arow = fi + ((b * HB + hp) * WBD + wp) * CF;
    for (int ch = 0; ch < CF; ch += 32) {
      Frag a;
      if (arow) {
        const u16* p = arow + ch + hi * 8;
        a.lo = *(const uint4*)(p);
        a.hi = *(const uint4*)(p + 16);
        if (ch + 32 < CF) __builtin_prefetch(arow + ch + 32, 0, 1);
      } else {
        a = fragzero();
      }
#pragma unroll
      for (int t = 0; t < 8; ++t) {
        Frag bfr = ldtile(wconv + (n0 + t * 16) * KCNV + tap * CF + ch, KCNV, lane);
        acc[t] = wmma16(a, bfr, acc[t]);
      }
    }
  }
  // transpose through LDS (stride 17: conflict-free) -> contiguous NCHW stores
#pragma unroll
  for (int t = 0; t < 8; ++t)
#pragma unroll
    for (int v = 0; v < 8; ++v)
      sm[(t * 16 + nl) * 17 + v + 8 * hi] = acc[t][v];
  __syncthreads();
  int vw = WBD - w0; if (vw > 16) vw = 16;
  int m = nl;
  for (int it = 0; it < 64; ++it) {
    int ol = it * 2 + hi;
    if (m < vw)
      out[((b * CF + n0 + ol) * HB + h) * WBD + w0 + m] = sm[ol * 17 + m];
  }
}

// per-channel sum/sumsq over (B,H,W) — deterministic block reduction
__global__ void stats_k(const float* __restrict__ c, float* __restrict__ stats) {
  int ch = blockIdx.x;
  int tid = threadIdx.x;
  float s = 0.f, s2 = 0.f;
  for (int b = 0; b < BB; ++b) {
    const float* p = c + (b * CF + ch) * HW;
    for (int i = tid; i < HW; i += 256) { float v = p[i]; s += v; s2 += v * v; }
  }
  __shared__ float rs[256], rq[256];
  rs[tid] = s; rq[tid] = s2;
  __syncthreads();
  for (int o = 128; o > 0; o >>= 1) {
    if (tid < o) { rs[tid] += rs[tid + o]; rq[tid] += rq[tid + o]; }
    __syncthreads();
  }
  if (tid == 0) { stats[ch] = rs[0]; stats[CF + ch] = rq[0]; }
}

__global__ void bnrelu_k(float* __restrict__ c, const float* __restrict__ stats,
                         const float* __restrict__ g, const float* __restrict__ b) {
  int i = blockIdx.x * blockDim.x + threadIdx.x;
  if (i >= BB * CF * HW) return;
  int ch = (i / HW) % CF;
  const float N = (float)(BB * HW);
  float mean = stats[ch] / N;
  float var = stats[CF + ch] / N - mean * mean;
  float sc = g[ch] * rsqrtf(var + 1e-5f);
  float shf = b[ch] - mean * sc;
  c[i] = fmaxf(c[i] * sc + shf, 0.f);
}

// ----------------------------- workspace layout ----------------------------
// (phase-based aliasing; total ~301 MB)
#define OFF_WB_IMG   ((size_t)0)
#define OFF_WB_RAD   ((size_t)65536)
#define OFF_WB_S1    ((size_t)98304)
#define OFF_WB_QKV   ((size_t)131072)
#define OFF_WB_OUT   ((size_t)229376)
#define OFF_WB_F1    ((size_t)262144)
#define OFF_WB_F2    ((size_t)393216)
#define OFF_WB_DELTA ((size_t)524288)
#define OFF_WB_GATE  ((size_t)557056)
#define OFF_WB_CONV  ((size_t)655360)
#define OFF_GIN      ((size_t)3309568)            // gin_pm bf16 [P][384]
#define OFF_FI       ((size_t)64749568)           // fi_pm bf16 [P][384]  (alias: seq_bf [P][256])
#define OFF_X        ((size_t)126189568)          // f32 [P][128] rad_seq -> x (in-place residual)
#define OFF_IMGS     ((size_t)167149568)          // f32 [P][128] img_seq -> delta_pm
#define OFF_SCORE    ((size_t)208109568)
#define OFF_IDX      ((size_t)208429568)
#define OFF_GP       ((size_t)208589568)          // qln bf16 | qh bf16 -> gate_pre f32
#define OFF_KVLN     ((size_t)249549568)
#define OFF_KH       ((size_t)259789568)
#define OFF_VH       ((size_t)270029568)
#define OFF_OB       ((size_t)280269568)          // hscore -> obuf -> x_bf
#define OFF_STAT     ((size_t)300749568)

extern "C" void kernel_launch(void* const* d_in, const int* in_sizes, int n_in,
                              void* d_out, int out_size, void* d_ws, size_t ws_size,
                              hipStream_t stream) {
  (void)in_sizes; (void)n_in; (void)out_size; (void)ws_size;
  const float* radar      = (const float*)d_in[0];
  const float* image      = (const float*)d_in[1];
  const float* img_proj_w = (const float*)d_in[2];
  const float* rad_proj_w = (const float*)d_in[3];
  const float* in_proj_w  = (const float*)d_in[4];
  const float* in_proj_b  = (const float*)d_in[5];
  const float* out_proj_w = (const float*)d_in[6];
  const float* out_proj_b = (const float*)d_in[7];
  const float* ln1q_g     = (const float*)d_in[8];
  const float* ln1q_b     = (const float*)d_in[9];
  const float* ln1kv_g    = (const float*)d_in[10];
  const float* ln1kv_b    = (const float*)d_in[11];
  const float* ln2_g      = (const float*)d_in[12];
  const float* ln2_b      = (const float*)d_in[13];
  const float* ffn_w1     = (const float*)d_in[14];
  const float* ffn_b1     = (const float*)d_in[15];
  const float* ffn_w2     = (const float*)d_in[16];
  const float* ffn_b2     = (const float*)d_in[17];
  const float* score_w1   = (const float*)d_in[18];
  const float* score_b1   = (const float*)d_in[19];
  const float* score_w2   = (const float*)d_in[20];
  const float* score_b2   = (const float*)d_in[21];
  const float* rad_delta_w= (const float*)d_in[22];
  const float* gamma      = (const float*)d_in[23];
  const float* gate_w     = (const float*)d_in[24];
  const float* gate_b     = (const float*)d_in[25];
  const float* fuse_w     = (const float*)d_in[26];
  const float* bn_g       = (const float*)d_in[27];
  const float* bn_b       = (const float*)d_in[28];
  float* out = (float*)d_out;

  char* ws = (char*)d_ws;
  u16*   wbImg  = (u16*)(ws + OFF_WB_IMG);
  u16*   wbRad  = (u16*)(ws + OFF_WB_RAD);
  u16*   wbS1   = (u16*)(ws + OFF_WB_S1);
  u16*   wbQKV  = (u16*)(ws + OFF_WB_QKV);
  u16*   wbOut  = (u16*)(ws + OFF_WB_OUT);
  u16*   wbF1   = (u16*)(ws + OFF_WB_F1);
  u16*   wbF2   = (u16*)(ws + OFF_WB_F2);
  u16*   wbDel  = (u16*)(ws + OFF_WB_DELTA);
  u16*   wbGate = (u16*)(ws + OFF_WB_GATE);
  u16*   wbConv = (u16*)(ws + OFF_WB_CONV);
  u16*   gin    = (u16*)(ws + OFF_GIN);
  u16*   fi     = (u16*)(ws + OFF_FI);
  u16*   seqbf  = (u16*)(ws + OFF_FI);                       // alias (phase 1-2)
  float* xbuf   = (float*)(ws + OFF_X);                      // rad_seq / x
  float* imgsf  = (float*)(ws + OFF_IMGS);                   // img_seq
  float* deltap = (float*)(ws + OFF_IMGS);                   // alias (phase 4)
  float* score  = (float*)(ws + OFF_SCORE);
  int*   idx    = (int*)(ws + OFF_IDX);
  u16*   qln    = (u16*)(ws + OFF_GP);
  u16*   qh     = (u16*)(ws + OFF_GP + 20480000);
  float* gatep  = (float*)(ws + OFF_GP);                     // alias (phase 4)
  u16*   kvln   = (u16*)(ws + OFF_KVLN);
  u16*   khb    = (u16*)(ws + OFF_KH);
  u16*   vhb    = (u16*)(ws + OFF_VH);
  u16*   hs     = (u16*)(ws + OFF_OB);                       // score hidden
  u16*   obuf   = (u16*)(ws + OFF_OB);                       // attention out
  u16*   xbf    = (u16*)(ws + OFF_OB);                       // x as bf16
  float* stats  = (float*)(ws + OFF_STAT);

  // phase 0: weight conversion + pixel-major transpose
  cvt_k<<<(CA*CI + 255) / 256, 256, 0, stream>>>(img_proj_w, wbImg, CA * CI);
  cvt_k<<<(CA*CR + 255) / 256, 256, 0, stream>>>(rad_proj_w, wbRad, CA * CR);
  cvt_k<<<(64*256 + 255) / 256, 256, 0, stream>>>(score_w1, wbS1, 64 * 256);
  cvt_k<<<(384*CA + 255) / 256, 256, 0, stream>>>(in_proj_w, wbQKV, 384 * CA);
  cvt_k<<<(CA*CA + 255) / 256, 256, 0, stream>>>(out_proj_w, wbOut, CA * CA);
  cvt_k<<<(512*CA + 255) / 256, 256, 0, stream>>>(ffn_w1, wbF1, 512 * CA);
  cvt_k<<<(CA*512 + 255) / 256, 256, 0, stream>>>(ffn_w2, wbF2, CA * 512);
  cvt_k<<<(CA*CA + 255) / 256, 256, 0, stream>>>(rad_delta_w, wbDel, CA * CA);
  cvt_k<<<(CR*CF + 255) / 256, 256, 0, stream>>>(gate_w, wbGate, CR * CF);
  convw_k<<<(CF*KCNV + 255) / 256, 256, 0, stream>>>(fuse_w, wbConv);
  pm_k<<<(BB*CF*HW + 255) / 256, 256, 0, stream>>>(radar, image, gin);

  // phase 1: projections (rad_seq, img_seq) f32 + bf16 concat copy
  gemm16<8, false><<<dim3(PIX/16, 1), 32, 0, stream>>>(gin, CF, wbRad, CR,
      nullptr, nullptr, 0, xbuf, CA, seqbf, 256);
  gemm16<8, false><<<dim3(PIX/16, 1), 32, 0, stream>>>(gin + CR, CF, wbImg, CI,
      nullptr, nullptr, 0, imgsf, CA, seqbf + 128, 256);

  // phase 2: score MLP + top-k
  gemm16<4, true><<<dim3(PIX/16, 1), 32, 0, stream>>>(seqbf, 256, wbS1, 256,
      score_b1, nullptr, 0, nullptr, 0, hs, 64);
  score2_k<<<(PIX + 255) / 256, 256, 0, stream>>>(hs, score_w2, score_b2, score);
  topk_k<<<ROWS, 256, 0, stream>>>(score, idx);

  // phase 3: attention block
  ln_rows_k<<<(PIX + 255) / 256, 256, 0, stream>>>(xbuf, ln1q_g, ln1q_b, qln, PIX);
  lnkv_k<<<(ROWS*KTOP + 255) / 256, 256, 0, stream>>>(imgsf, idx, ln1kv_g, ln1kv_b, kvln);
  gemm16<8, false><<<dim3(PIX/16, 1), 32, 0, stream>>>(qln, CA, wbQKV, CA,
      in_proj_b, nullptr, 0, nullptr, 0, qh, CA);
  gemm16<8, false><<<dim3(ROWS*KTOP/16, 1), 32, 0, stream>>>(kvln, CA, wbQKV + CA*CA, CA,
      in_proj_b + CA, nullptr, 0, nullptr, 0, khb, CA);
  gemm16<8, false><<<dim3(ROWS*KTOP/16, 1), 32, 0, stream>>>(kvln, CA, wbQKV + 2*CA*CA, CA,
      in_proj_b + 2*CA, nullptr, 0, nullptr, 0, vhb, CA);
  attn_k<<<ROWS * NHD, 32, 0, stream>>>(qh, khb, vhb, obuf);
  gemm16<8, false><<<dim3(PIX/16, 1), 32, 0, stream>>>(obuf, CA, wbOut, CA,
      out_proj_b, xbuf, CA, xbuf, CA, nullptr, 0);      // x = rad_seq + o@Wo^T+b
  ffn_k<<<PIX/16, 32, 0, stream>>>(xbuf, xbf, ln2_g, ln2_b, wbF1, ffn_b1, wbF2, ffn_b2);

  // phase 4: delta / gate / radar enhancement -> fi_pm
  gemm16<8, false><<<dim3(PIX/16, 1), 32, 0, stream>>>(xbf, CA, wbDel, CA,
      nullptr, nullptr, 0, deltap, CA, nullptr, 0);
  gemm16<8, false><<<dim3(PIX/16, 1), 32, 0, stream>>>(gin, CF, wbGate, CF,
      gate_b, nullptr, 0, gatep, CR, nullptr, 0);
  enhance_k<<<(PIX + 255) / 256, 256, 0, stream>>>(radar, gatep, deltap, gamma, fi);
  copyimg_k<<<(PIX*CI + 255) / 256, 256, 0, stream>>>(gin, fi);

  // phase 5: 3x3 fuse conv + batch-norm-from-data + relu
  conv_k<<<dim3(BB*HB*13, 3), 32, 0, stream>>>(fi, wbConv, out);
  stats_k<<<CF, 256, 0, stream>>>(out, stats);
  bnrelu_k<<<(BB*CF*HW + 255) / 256, 256, 0, stream>>>(out, stats, bn_g, bn_b);
}
